// ListSNNMulti_32676111188714
// MI455X (gfx1250) — matrix operand order repaired and checked by reading, not compiled, not run on hardware
//
#include <hip/hip_runtime.h>
#include <math.h>

// ---------------------------------------------------------------------------
// SNN (SRM/LIF, soft reset) on MI455X / gfx1250.
// Phase 1: time-parallel GEMM1 (x @ W1^T for all t), 2x2 register-blocked
//          v_wmma_f32_16x16x32_f16 (2 b128 loads per WMMA).
// Phase 2: 32 persistent workgroups scan over T, all LIF state in LDS
//          (256KB/WGP), GEMM2/GEMM3 per step via WMMA with B-frag reuse.
// Workspace requirement: ~216 MB (xh + packed f16 weights + C1 f32).
// ---------------------------------------------------------------------------

typedef __attribute__((ext_vector_type(16))) _Float16 v16h;
typedef __attribute__((ext_vector_type(8)))  _Float16 v8h;
typedef __attribute__((ext_vector_type(8)))  float    v8f;

#define M_MODELS 4
#define B_SZ     256
#define T_STEPS  100
#define DIN      784
#define DH       400
#define DOUT     10
#define PK_IN    800   // DIN padded to multiple of 32
#define PK_H     416   // DH  padded to multiple of 32 (K dim of GEMM2/3)
#define PO1      416   // DH  padded to multiple of 32 (N dim of GEMM1 / C1 width)
#define PO3      16    // DOUT padded to 16
#define AI       0.75f
#define AV       0.9f
#define VTH      1.0f

union FragH { v16h v; v8h h[2]; };

// --------------------------- pack kernels ----------------------------------

__global__ void pack_x_kernel(const float* __restrict__ x, _Float16* __restrict__ xh) {
    long idx = (long)blockIdx.x * blockDim.x + threadIdx.x;       // over B*T*PK_IN
    if (idx >= (long)B_SZ * T_STEPS * PK_IN) return;
    int i  = (int)(idx % PK_IN);
    long bt = idx / PK_IN;                                        // b*T + t
    xh[idx] = (i < DIN) ? (_Float16)x[bt * DIN + i] : (_Float16)0.0f;
}

__global__ void pack_w_kernel(const float* __restrict__ w, _Float16* __restrict__ wh,
                              int O, int I, int OP, int IP) {
    long idx = (long)blockIdx.x * blockDim.x + threadIdx.x;       // over M*OP*IP
    if (idx >= (long)M_MODELS * OP * IP) return;
    int i = (int)(idx % IP);
    int o = (int)((idx / IP) % OP);
    int m = (int)(idx / ((long)OP * IP));
    float v = (o < O && i < I) ? w[((long)m * O + o) * I + i] : 0.0f;
    wh[idx] = (_Float16)v;
}

// --------------------------- WMMA frag loaders -----------------------------
// 16-bit A 16x32 (MxK): lane l<16 holds row M=l, K=K0..K0+15 contiguous;
//                       lane l>=16 holds row M=l-16, K=K0+16..K0+31.
__device__ __forceinline__ v16h load_a_rowmajor(const _Float16* __restrict__ base,
                                                long row_stride, int k0, int lane) {
    const int hi  = lane >> 4;
    const int l15 = lane & 15;
    const _Float16* p = base + (long)l15 * row_stride + k0 + hi * 16;
    FragH a;
    a.h[0] = *(const v8h*)p;
    a.h[1] = *(const v8h*)(p + 8);
    return a.v;
}

// 16-bit B 32x16 (KxN): lane l<16 holds col N=l, K=k0..k0+15 contiguous;
//                       lane l>=16 holds col N=l-16, K=k0+16..k0+31.
__device__ __forceinline__ v16h load_b_rowmajorT(const _Float16* __restrict__ w,
                                                 int oc, int k0, int row_stride, int lane) {
    const int hi  = lane >> 4;
    const int l15 = lane & 15;
    const _Float16* p = w + (long)(oc + l15) * row_stride + k0 + hi * 16;
    FragH b;
    b.h[0] = *(const v8h*)p;
    b.h[1] = *(const v8h*)(p + 8);
    return b.v;
}

#define WMMA_F16(A, B, C) \
    __builtin_amdgcn_wmma_f32_16x16x32_f16(false, (A), false, (B), (short)0, (C), false, false)

// --------------------------- phase 1: GEMM1 --------------------------------
// C1[m][b][t][o] = sum_i x[b][t][i] * W1[m][o][i], all (m,t,b,o) in parallel.
// 2x2 register blocking: each wave computes a 32b x 32o block.
// Blocks: m(4) x t(100) x bblk(8) x oblk(13) = 41600 waves; 8 waves/block.
__global__ void gemm1_kernel(const _Float16* __restrict__ xh,
                             const _Float16* __restrict__ w1h,
                             float* __restrict__ c1) {
    const int lane = threadIdx.x & 31;
    const int wave = threadIdx.x >> 5;
    long tile = (long)blockIdx.x * 8 + wave;
    const int ob = (int)(tile % 13); tile /= 13;
    const int bb = (int)(tile % 8);  tile /= 8;
    const int t  = (int)(tile % T_STEPS); tile /= T_STEPS;
    const int m  = (int)tile;
    const int b0 = bb * 32, oc = ob * 32;

    const _Float16* xbase = xh + ((long)b0 * T_STEPS + t) * PK_IN;
    const long      xstr  = (long)T_STEPS * PK_IN;
    const _Float16* wbase = w1h + (long)m * PO1 * PK_IN;

    v8f acc00 = {}, acc01 = {}, acc10 = {}, acc11 = {};
    for (int k0 = 0; k0 < PK_IN; k0 += 32) {
        v16h a0 = load_a_rowmajor(xbase, xstr, k0, lane);
        v16h a1 = load_a_rowmajor(xbase + 16 * xstr, xstr, k0, lane);
        v16h bf0 = load_b_rowmajorT(wbase, oc,      k0, PK_IN, lane);
        v16h bf1 = load_b_rowmajorT(wbase, oc + 16, k0, PK_IN, lane);
        acc00 = WMMA_F16(a0, bf0, acc00);
        acc01 = WMMA_F16(a0, bf1, acc01);
        acc10 = WMMA_F16(a1, bf0, acc10);
        acc11 = WMMA_F16(a1, bf1, acc11);
    }
    // C/D layout: M = j + 8*(lane>=16), N = lane&15
    const int hi = lane >> 4, l15 = lane & 15;
    for (int j = 0; j < 8; ++j) {
        const int br = hi * 8 + j;
        long base0 = (((long)m * B_SZ + b0 + br) * T_STEPS + t) * PO1;
        long base1 = (((long)m * B_SZ + b0 + 16 + br) * T_STEPS + t) * PO1;
        c1[base0 + oc + l15]      = acc00[j];
        c1[base0 + oc + 16 + l15] = acc01[j];
        c1[base1 + oc + l15]      = acc10[j];
        c1[base1 + oc + 16 + l15] = acc11[j];
    }
}

// --------------------------- phase 2: scan ---------------------------------
// 32 blocks (m x 8 batch-tiles of 32 rows) x 512 threads (16 waves).
__global__ void scan_kernel(const float* __restrict__ c1,
                            const _Float16* __restrict__ w2h,
                            const _Float16* __restrict__ w3h,
                            const float* __restrict__ b1g,
                            const float* __restrict__ b2g,
                            const float* __restrict__ b3g,
                            float* __restrict__ out, float beta) {
    extern __shared__ char smem[];
    float*    i1  = (float*)smem;                       // 32*400 f32
    float*    v1  = i1 + 32 * DH;
    float*    i2  = v1 + 32 * DH;
    float*    v2  = i2 + 32 * DH;
    float*    i3  = v2 + 32 * DH;                       // 32*16 f32
    float*    v3  = i3 + 32 * PO3;
    _Float16* s1h = (_Float16*)(v3 + 32 * PO3);         // 32*416 f16
    _Float16* s2h = s1h + 32 * PK_H;                    // 32*416 f16
    // total = 4*4*12800 + 2*4*512 + 2*2*13312 = 262144 bytes

    const int m  = blockIdx.x >> 3;
    const int b0 = (blockIdx.x & 7) * 32;
    const int lane = threadIdx.x & 31;
    const int wave = threadIdx.x >> 5;
    const int hi = lane >> 4, l15 = lane & 15;

    // zero all LDS (states + spike pads)
    {
        int* zp = (int*)smem;
        for (int idx = threadIdx.x; idx < 262144 / 4; idx += 512) zp[idx] = 0;
    }
    __syncthreads();

    const _Float16* w2m = w2h + (long)m * DH * PK_H;
    const _Float16* w3m = w3h + (long)m * PO3 * PK_H;

    for (int t = 0; t < T_STEPS; ++t) {
        // ---- layer 1: consume precomputed cur1, LIF update, emit s1 (f16) ----
        for (int idx = threadIdx.x; idx < 32 * DH; idx += 512) {
            const int b = idx / DH, o = idx % DH;
            const float cur = c1[(((long)m * B_SZ + b0 + b) * T_STEPS + t) * PO1 + o];
            const float ii  = AI * i1[idx] + cur;
            float       vv  = AV * v1[idx] + beta * ii + beta * b1g[m * DH + o];
            const float s   = (vv >= VTH) ? 1.0f : 0.0f;
            i1[idx] = ii;
            v1[idx] = vv - s * VTH;
            s1h[b * PK_H + o] = (_Float16)s;
        }
        __syncthreads();

        // ---- layer 2: GEMM (s1 @ W2^T) + fused LIF ----
        // 25 o-tiles; each unit reuses the W2 B-frag for both 16-row b-halves.
        for (int unit = wave; unit < 25; unit += 16) {
            const int oc = unit * 16;
            v8f accA = {}, accB = {};
            for (int k0 = 0; k0 < PK_H; k0 += 32) {
                v16h a0 = load_a_rowmajor(s1h, PK_H, k0, lane);
                v16h a1 = load_a_rowmajor(s1h + 16 * PK_H, PK_H, k0, lane);
                const _Float16* pb = w2m + (long)(oc + l15) * PK_H + k0 + hi * 16;
                __builtin_prefetch(pb + 32, 0, 1);      // global_prefetch next K slab
                FragH bf; bf.h[0] = *(const v8h*)pb; bf.h[1] = *(const v8h*)(pb + 8);
                accA = WMMA_F16(a0, bf.v, accA);
                accB = WMMA_F16(a1, bf.v, accB);
            }
            for (int half = 0; half < 2; ++half) {
                const v8f acc = half ? accB : accA;
                for (int j = 0; j < 8; ++j) {
                    const int row = half * 16 + hi * 8 + j;   // local batch row 0..31
                    const int col = oc + l15;                 // hidden unit
                    const int idx = row * DH + col;
                    const float ii = AI * i2[idx] + acc[j];
                    float       vv = AV * v2[idx] + beta * ii + beta * b2g[m * DH + col];
                    const float s  = (vv >= VTH) ? 1.0f : 0.0f;
                    i2[idx] = ii;
                    v2[idx] = vv - s * VTH;
                    s2h[row * PK_H + col] = (_Float16)s;
                }
            }
        }
        __syncthreads();

        // ---- layer 3: GEMM (s2 @ W3^T, O padded to 16) + LIF + emit spikes ----
        if (wave < 2) {
            const int rb = wave * 16;
            v8f acc = {};
            for (int k0 = 0; k0 < PK_H; k0 += 32) {
                v16h a = load_a_rowmajor(s2h + (long)rb * PK_H, PK_H, k0, lane);
                v16h b = load_b_rowmajorT(w3m, 0, k0, PK_H, lane);
                acc = WMMA_F16(a, b, acc);
            }
            for (int j = 0; j < 8; ++j) {
                const int row = rb + hi * 8 + j;        // local batch row
                const int col = l15;                    // output unit (0..15)
                const int idx = row * PO3 + col;
                const float bias = (col < DOUT) ? b3g[m * DOUT + col] : 0.0f;
                const float ii = AI * i3[idx] + acc[j];
                float       vv = AV * v3[idx] + beta * ii + beta * bias;
                const float s  = (vv >= VTH) ? 1.0f : 0.0f;
                i3[idx] = ii;
                v3[idx] = vv - s * VTH;
                if (col < DOUT) {
                    out[(((long)m * B_SZ + b0 + row) * T_STEPS + t) * DOUT + col] = s;
                }
            }
        }
        __syncthreads();
    }
}

// --------------------------- host launch -----------------------------------

extern "C" void kernel_launch(void* const* d_in, const int* in_sizes, int n_in,
                              void* d_out, int out_size, void* d_ws, size_t ws_size,
                              hipStream_t stream) {
    const float* x  = (const float*)d_in[0];   // [B,T,784]
    const float* W1 = (const float*)d_in[1];   // [M,400,784]
    const float* W2 = (const float*)d_in[2];   // [M,400,400]
    const float* W3 = (const float*)d_in[3];   // [M,10,400]
    const float* b1 = (const float*)d_in[4];   // [M,400]
    const float* b2 = (const float*)d_in[5];   // [M,400]
    const float* b3 = (const float*)d_in[6];   // [M,10]
    float* out = (float*)d_out;                // [M,B,T,10]

    char* ws = (char*)d_ws;
    size_t off = 0;
    _Float16* xh  = (_Float16*)(ws + off); off += (size_t)B_SZ * T_STEPS * PK_IN * 2;   //  40.96 MB
    _Float16* w1h = (_Float16*)(ws + off); off += (size_t)M_MODELS * PO1 * PK_IN * 2;   //   2.66 MB
    _Float16* w2h = (_Float16*)(ws + off); off += (size_t)M_MODELS * DH * PK_H * 2;     //   1.33 MB
    _Float16* w3h = (_Float16*)(ws + off); off += (size_t)M_MODELS * PO3 * PK_H * 2;    //   0.05 MB
    float*    c1  = (float*)(ws + off);    off += (size_t)M_MODELS * B_SZ * T_STEPS * PO1 * 4; // 170.39 MB

    // beta_v = 1 / max_t eps[t],  eps[t] = (ai^(t+1) - av^(t+1)) / (ai - av)
    double mx = 0.0;
    for (int t = 0; t < T_STEPS; ++t) {
        double e = (pow(0.75, t + 1) - pow(0.9, t + 1)) / (0.75 - 0.9);
        if (e > mx) mx = e;
    }
    const float beta = (float)(1.0 / mx);

    {   // pack inputs/weights to padded f16
        long n = (long)B_SZ * T_STEPS * PK_IN;
        pack_x_kernel<<<(int)((n + 255) / 256), 256, 0, stream>>>(x, xh);
        long n1 = (long)M_MODELS * PO1 * PK_IN;
        pack_w_kernel<<<(int)((n1 + 255) / 256), 256, 0, stream>>>(W1, w1h, DH, DIN, PO1, PK_IN);
        long n2 = (long)M_MODELS * DH * PK_H;
        pack_w_kernel<<<(int)((n2 + 255) / 256), 256, 0, stream>>>(W2, w2h, DH, DH, DH, PK_H);
        long n3 = (long)M_MODELS * PO3 * PK_H;
        pack_w_kernel<<<(int)((n3 + 255) / 256), 256, 0, stream>>>(W3, w3h, DOUT, DH, PO3, PK_H);
    }

    // phase 1: 41600 wave-tiles / 8 waves = 5200 blocks
    gemm1_kernel<<<5200, 256, 0, stream>>>(xh, w1h, c1);

    // phase 2: 32 persistent workgroups, 512 threads, 256 KB dynamic LDS each
    scan_kernel<<<32, 512, 262144, stream>>>(c1, w2h, w3h, b1, b2, b3, out, beta);

    (void)in_sizes; (void)n_in; (void)out_size; (void)ws_size;
}
